// GraphAttentionHead_15135464751289
// MI455X (gfx1250) — matrix operand order repaired and checked by reading, not compiled, not run on hardware
//
#include <hip/hip_runtime.h>

typedef __attribute__((ext_vector_type(16))) _Float16 v16h;
typedef __attribute__((ext_vector_type(8)))  float    v8f;

#define B_ 8
#define N_ 2048
#define D_ 128
#define NEG_SLOPE 0.2f

// ---------------------------------------------------------------------------
// Workspace layout (bytes):
//   [0, 32768)            : WT_f16  : W transposed+converted, WT[d][k], 128x128 f16
//   [32768, +4MiB)        : WhT_f16 : WhT[b][d][n], f16 (B-operand layout for pass 2)
//   then 4x (B_*N_) f32   : s1 (Wh@a[:D]), s2 (Wh@a[D:]), m (row max), l (row sumexp)
//   then up to 3x out-size f32 partial accumulators for split-j (if ws allows)
// ---------------------------------------------------------------------------

// Kernel 0: W (f32, k-major) -> WT (f16, d-major)
__global__ void __launch_bounds__(256)
gat_k0_wt(const float* __restrict__ W, _Float16* __restrict__ WT) {
    int idx = blockIdx.x * 256 + threadIdx.x;     // 0 .. 16383
    int k = idx >> 7;
    int d = idx & 127;
    WT[d * D_ + k] = (_Float16)W[idx];
}

// Kernel 1: Wh = x @ W via WMMA f16; store WhT f16 (transposed) + s1, s2.
// One wave (32 threads) per 16-row tile. 1024 blocks.
__global__ void __launch_bounds__(32)
gat_k1_wh(const float* __restrict__ x, const _Float16* __restrict__ WT,
          const float* __restrict__ a, _Float16* __restrict__ WhT,
          float* __restrict__ s1, float* __restrict__ s2) {
    const int t  = blockIdx.x;            // tile id
    const int b  = t >> 7;                // 128 tiles per batch
    const int n0 = (t & 127) << 4;
    const int lane = threadIdx.x;
    const int r = lane & 15;              // row within tile (A), column (B/C)
    const int g = lane >> 4;              // half-wave

    const float* xrow = x + ((size_t)(b * N_) + n0 + r) * D_;

    v8f acc[8];
    v8f zero = {};
#pragma unroll
    for (int dt = 0; dt < 8; ++dt) acc[dt] = zero;

#pragma unroll
    for (int kc = 0; kc < D_; kc += 32) {
        // --- A fragment: x[n0+r][kc + {8g..8g+7, 16+8g..16+8g+7}] -> f16 ---
        const float4* pa = (const float4*)(xrow + kc + 8 * g);
        float4 f0 = pa[0], f1 = pa[1];
        const float4* pb = (const float4*)(xrow + kc + 16 + 8 * g);
        float4 f2 = pb[0], f3 = pb[1];
        v16h af;
        af[0]  = (_Float16)f0.x; af[1]  = (_Float16)f0.y;
        af[2]  = (_Float16)f0.z; af[3]  = (_Float16)f0.w;
        af[4]  = (_Float16)f1.x; af[5]  = (_Float16)f1.y;
        af[6]  = (_Float16)f1.z; af[7]  = (_Float16)f1.w;
        af[8]  = (_Float16)f2.x; af[9]  = (_Float16)f2.y;
        af[10] = (_Float16)f2.z; af[11] = (_Float16)f2.w;
        af[12] = (_Float16)f3.x; af[13] = (_Float16)f3.y;
        af[14] = (_Float16)f3.z; af[15] = (_Float16)f3.w;

#pragma unroll
        for (int dt = 0; dt < 8; ++dt) {
            // B fragment: WT[dt*16 + r][kc + {8g..8g+7, 16+8g..16+8g+7}]
            union { int4 q[2]; v16h v; } bu;
            const int4* bp = (const int4*)(WT + (size_t)(dt * 16 + r) * D_ + kc + 8 * g);
            bu.q[0] = bp[0];
            bu.q[1] = bp[2];
            acc[dt] = __builtin_amdgcn_wmma_f32_16x16x32_f16(
                false, af, false, bu.v, (short)0, acc[dt], false, false);
        }
    }

    // Store WhT f16 and accumulate s1/s2 partials.
    float p1v[8], p2v[8];
#pragma unroll
    for (int v = 0; v < 8; ++v) { p1v[v] = 0.f; p2v[v] = 0.f; }

#pragma unroll
    for (int dt = 0; dt < 8; ++dt) {
        int d = dt * 16 + r;
        float a1 = a[d];
        float a2 = a[D_ + d];
        union { int4 q; _Float16 h[8]; } st;
#pragma unroll
        for (int v = 0; v < 8; ++v) {
            float w = acc[dt][v];            // Wh[n0 + v + 8g][d]
            st.h[v] = (_Float16)w;
            p1v[v] += w * a1;
            p2v[v] += w * a2;
        }
        *(int4*)(WhT + ((size_t)b * D_ + d) * N_ + n0 + 8 * g) = st.q;
    }

    // Butterfly reduce over the 16 lanes of each half-wave (columns d).
#pragma unroll
    for (int mask = 1; mask < 16; mask <<= 1) {
#pragma unroll
        for (int v = 0; v < 8; ++v) {
            p1v[v] += __shfl_xor(p1v[v], mask, 32);
            p2v[v] += __shfl_xor(p2v[v], mask, 32);
        }
    }
    if (r == 0) {
        int base = b * N_ + n0 + 8 * g;
#pragma unroll
        for (int v = 0; v < 8; ++v) {
            s1[base + v] = p1v[v];
            s2[base + v] = p2v[v];
        }
    }
}

// Kernel 2: per-row masked max and sum-of-exp. One wave per row, 8 rows/block.
__global__ void __launch_bounds__(256)
gat_k2_ml(const int* __restrict__ adj, const float* __restrict__ s1,
          const float* __restrict__ s2, float* __restrict__ mrow,
          float* __restrict__ lrow) {
    const int wave = threadIdx.x >> 5;
    const int lane = threadIdx.x & 31;
    const int row  = blockIdx.x * 8 + wave;       // global row id in [0, B_*N_)
    const int b    = row >> 11;                   // N_ == 2048

    const float si = s1[row];
    const int*   arow = adj + (size_t)row * N_;
    const float* sj   = s2 + b * N_;

    const float NEG_BIG = -__builtin_inff();

    // Sweep 1: masked row max (branchless).
    float m = NEG_BIG;
    for (int t = 0; t < N_ / 128; ++t) {
        int j = t * 128 + lane * 4;
        int4   av = *(const int4*)(arow + j);
        float4 sv = *(const float4*)(sj + j);
        float e0 = si + sv.x; e0 = e0 >= 0.f ? e0 : NEG_SLOPE * e0;
        float e1 = si + sv.y; e1 = e1 >= 0.f ? e1 : NEG_SLOPE * e1;
        float e2 = si + sv.z; e2 = e2 >= 0.f ? e2 : NEG_SLOPE * e2;
        float e3 = si + sv.w; e3 = e3 >= 0.f ? e3 : NEG_SLOPE * e3;
        m = fmaxf(m, av.x ? e0 : NEG_BIG);
        m = fmaxf(m, av.y ? e1 : NEG_BIG);
        m = fmaxf(m, av.z ? e2 : NEG_BIG);
        m = fmaxf(m, av.w ? e3 : NEG_BIG);
    }
#pragma unroll
    for (int mask = 1; mask < 32; mask <<= 1)
        m = fmaxf(m, __shfl_xor(m, mask, 32));

    // Sweep 2: sum of exp(e - m) over edges (adj row hits L2).
    float l0 = 0.f, l1 = 0.f, l2 = 0.f, l3 = 0.f;
    for (int t = 0; t < N_ / 128; ++t) {
        int j = t * 128 + lane * 4;
        int4   av = *(const int4*)(arow + j);
        float4 sv = *(const float4*)(sj + j);
        float e0 = si + sv.x; e0 = e0 >= 0.f ? e0 : NEG_SLOPE * e0;
        float e1 = si + sv.y; e1 = e1 >= 0.f ? e1 : NEG_SLOPE * e1;
        float e2 = si + sv.z; e2 = e2 >= 0.f ? e2 : NEG_SLOPE * e2;
        float e3 = si + sv.w; e3 = e3 >= 0.f ? e3 : NEG_SLOPE * e3;
        if (av.x) l0 += __expf(e0 - m);
        if (av.y) l1 += __expf(e1 - m);
        if (av.z) l2 += __expf(e2 - m);
        if (av.w) l3 += __expf(e3 - m);
    }
    float l = (l0 + l1) + (l2 + l3);
#pragma unroll
    for (int mask = 1; mask < 32; mask <<= 1)
        l += __shfl_xor(l, mask, 32);

    if (lane == 0) { mrow[row] = m; lrow[row] = l; }
}

// Kernel 3: fused masked-softmax * Wh via WMMA over j in [j0, j1).
// One wave per 16-row tile; writes UNNORMALIZED partial sums to dst.
__global__ void __launch_bounds__(32)
gat_k3_part(const int* __restrict__ adj, const _Float16* __restrict__ WhT,
            const float* __restrict__ s1, const float* __restrict__ s2,
            const float* __restrict__ mrow, float* __restrict__ dst,
            int j0, int j1) {
    const int t  = blockIdx.x;
    const int b  = t >> 7;
    const int i0 = (t & 127) << 4;
    const int lane = threadIdx.x;
    const int r = lane & 15;
    const int g = lane >> 4;

    const int gi = b * N_ + i0 + r;           // this lane's A-fragment row
    const float si = s1[gi];
    const float em = mrow[gi];
    const int*      arow = adj + (size_t)gi * N_;
    const float*    sj   = s2 + b * N_;
    const _Float16* whb  = WhT + (size_t)b * D_ * N_;

    v8f acc[8];
    v8f zero = {};
#pragma unroll
    for (int dt = 0; dt < 8; ++dt) acc[dt] = zero;

    for (int jc = j0; jc < j1; jc += 32) {
        const int ja = jc + 8 * g;
        const int jb = jc + 16 + 8 * g;
        int4 a0 = *(const int4*)(arow + ja);
        int4 a1 = *(const int4*)(arow + ja + 4);
        int4 a2 = *(const int4*)(arow + jb);
        int4 a3 = *(const int4*)(arow + jb + 4);
        float4 v0 = *(const float4*)(sj + ja);
        float4 v1 = *(const float4*)(sj + ja + 4);
        float4 v2 = *(const float4*)(sj + jb);
        float4 v3 = *(const float4*)(sj + jb + 4);

        if (jc + 64 < j1) __builtin_prefetch(arow + jc + 64, 0, 0);

        int   avi[16] = {a0.x, a0.y, a0.z, a0.w, a1.x, a1.y, a1.z, a1.w,
                         a2.x, a2.y, a2.z, a2.w, a3.x, a3.y, a3.z, a3.w};
        float svf[16] = {v0.x, v0.y, v0.z, v0.w, v1.x, v1.y, v1.z, v1.w,
                         v2.x, v2.y, v2.z, v2.w, v3.x, v3.y, v3.z, v3.w};

        // A fragment: unnormalized alpha (f16), masked.
        v16h af;
#pragma unroll
        for (int q = 0; q < 16; ++q) {
            float e = si + svf[q];
            e = e >= 0.f ? e : NEG_SLOPE * e;
            float p = avi[q] ? __expf(e - em) : 0.f;
            af[q] = (_Float16)p;
        }

#pragma unroll
        for (int dt = 0; dt < 8; ++dt) {
            // B fragment: WhT[dt*16 + r][jc + {8g..8g+7, 16+8g..16+8g+7}]
            union { int4 q[2]; v16h v; } bu;
            const int4* bp = (const int4*)(whb + (size_t)(dt * 16 + r) * N_ + jc + 8 * g);
            bu.q[0] = bp[0];
            bu.q[1] = bp[2];
            acc[dt] = __builtin_amdgcn_wmma_f32_16x16x32_f16(
                false, af, false, bu.v, (short)0, acc[dt], false, false);
        }
    }

    // Store raw partial (normalization happens in k4).
#pragma unroll
    for (int dt = 0; dt < 8; ++dt) {
        int d = dt * 16 + r;
#pragma unroll
        for (int v = 0; v < 8; ++v) {
            dst[((size_t)(b * N_ + i0 + 8 * g + v)) * D_ + d] = acc[dt][v];
        }
    }
}

// Kernel 4: out = (out + p1 + p2 + p3) * (1/l[row]); fixed summation order.
__global__ void __launch_bounds__(256)
gat_k4_norm(float* __restrict__ out, const float* __restrict__ p1,
            const float* __restrict__ p2, const float* __restrict__ p3,
            const float* __restrict__ lrow, int nseg) {
    size_t base = ((size_t)blockIdx.x * 256 + threadIdx.x) * 4;
    int row = (int)(base >> 7);                  // /D_
    float4 v = *(float4*)(out + base);
    if (nseg > 1) {
        float4 q = *(const float4*)(p1 + base);
        v.x += q.x; v.y += q.y; v.z += q.z; v.w += q.w;
    }
    if (nseg > 2) {
        float4 q = *(const float4*)(p2 + base);
        v.x += q.x; v.y += q.y; v.z += q.z; v.w += q.w;
    }
    if (nseg > 3) {
        float4 q = *(const float4*)(p3 + base);
        v.x += q.x; v.y += q.y; v.z += q.z; v.w += q.w;
    }
    float linv = 1.f / lrow[row];
    v.x *= linv; v.y *= linv; v.z *= linv; v.w *= linv;
    *(float4*)(out + base) = v;
}

extern "C" void kernel_launch(void* const* d_in, const int* in_sizes, int n_in,
                              void* d_out, int out_size, void* d_ws, size_t ws_size,
                              hipStream_t stream) {
    const float* x   = (const float*)d_in[0];   // (B,N,D) f32
    const int*   adj = (const int*)d_in[1];     // (B,N,N) i32
    const float* W   = (const float*)d_in[2];   // (D,D)   f32
    const float* a   = (const float*)d_in[3];   // (2D,)   f32
    float* out = (float*)d_out;                 // (B,N,D) f32

    char* ws = (char*)d_ws;
    const size_t whtBytes = (size_t)B_ * D_ * N_ * 2;      // 4 MiB
    const size_t vecBytes = (size_t)B_ * N_ * 4;           // 64 KiB each
    _Float16* WT  = (_Float16*)ws;                         // 32 KiB
    _Float16* WhT = (_Float16*)(ws + 32768);
    float* s1 = (float*)(ws + 32768 + whtBytes);
    float* s2 = s1 + B_ * N_;
    float* mr = s2 + B_ * N_;
    float* lr = mr + B_ * N_;

    const size_t baseBytes = 32768 + whtBytes + 4 * vecBytes;
    const size_t outElems  = (size_t)B_ * N_ * D_;
    const size_t outBytes  = outElems * 4;                 // 8 MiB
    float* p1 = (float*)(ws + baseBytes);
    float* p2 = p1 + outElems;
    float* p3 = p2 + outElems;

    // Split the j-range into S segments for occupancy (deterministic: fixed
    // segment boundaries, partials summed in fixed order in k4).
    int S = 1;
    if (ws_size >= baseBytes + 3 * outBytes)      S = 4;
    else if (ws_size >= baseBytes + 1 * outBytes) S = 2;

    gat_k0_wt <<<(D_ * D_) / 256, 256, 0, stream>>>(W, WT);
    gat_k1_wh <<<(B_ * N_) / 16, 32, 0, stream>>>(x, WT, a, WhT, s1, s2);
    gat_k2_ml <<<(B_ * N_) / 8, 256, 0, stream>>>(adj, s1, s2, mr, lr);

    const int jseg = N_ / S;
    float* dsts[4] = {out, p1, p2, p3};
    for (int s = 0; s < S; ++s) {
        gat_k3_part<<<(B_ * N_) / 16, 32, 0, stream>>>(
            adj, WhT, s1, s2, mr, dsts[s], s * jseg, (s + 1) * jseg);
    }
    gat_k4_norm<<<(int)(outElems / 4 / 256), 256, 0, stream>>>(out, p1, p2, p3, lr, S);
}
